// DMPNN_16913581211836
// MI455X (gfx1250) — matrix operand order repaired and checked by reading, not compiled, not run on hardware
//
#include <hip/hip_runtime.h>
#include <hip/hip_bf16.h>

#define DIM 64

typedef __attribute__((ext_vector_type(16))) __bf16 v16bf;
typedef __attribute__((ext_vector_type(8)))  float  v8f;

// f32 -> bf16; backend emits native v_cvt_pk_bf16_f32 on gfx1250
__device__ __forceinline__ __bf16 f2bf(float x) { return static_cast<__bf16>(x); }

// 32-bit-offset addressing helpers: keep the base uniform (SGPR pair) and the
// offset a 32-bit byte value so stores/loads lower to saddr + voffset + imm.
__device__ __forceinline__ float ld_off(const float* base, unsigned boff) {
  return *reinterpret_cast<const float*>(reinterpret_cast<const char*>(base) + boff);
}
__device__ __forceinline__ void st_off(float* base, unsigned boff, float v) {
  *reinterpret_cast<float*>(reinterpret_cast<char*>(base) + boff) = v;
}

// A-matrix (16x32 bf16) per-lane fragment from an f32 row pointer.
// Lane layout (ISA 7.12.2, 16-bit A 16x32): lanes 0-15 hold K {0..7, 16..23},
// lanes 16-31 hold K {8..15, 24..31}.
__device__ __forceinline__ v16bf make_a_frag(const float* __restrict__ row, int k0, int half) {
  v16bf a;
  const float* p0 = row + k0 + half * 8;
  const float* p1 = p0 + 16;
#pragma unroll
  for (int j = 0; j < 8; ++j) {
    a[j]     = f2bf(p0[j]);
    a[j + 8] = f2bf(p1[j]);
  }
  return a;
}

// Same, but element = agg[src][k] - h[rev][k]  (the directed message)
__device__ __forceinline__ v16bf make_m_frag(const float* __restrict__ ar,
                                             const float* __restrict__ hr,
                                             int k0, int half) {
  v16bf a;
  int b0 = k0 + half * 8;
  int b1 = b0 + 16;
#pragma unroll
  for (int j = 0; j < 8; ++j) {
    a[j]     = f2bf(ar[b0 + j] - hr[b0 + j]);
    a[j + 8] = f2bf(ar[b1 + j] - hr[b1 + j]);
  }
  return a;
}

// Stage W (Ktot x 64, f32 row-major) into LDS as ready-to-load B fragments:
// sB[((kt*4+nt)*32 + lane)*16 + j] = bf16(W[kt*32 + (lane>>4)*8 + (j&7) + (j>>3)*16][nt*16 + (lane&15)])
template <int KT>
__device__ __forceinline__ void stage_w(const float* __restrict__ W, __bf16* sB) {
  const int total = KT * 4 * 32 * 16;
  for (int i = threadIdx.x; i < total; i += blockDim.x) {
    int j    = i & 15;
    int lane = (i >> 4) & 31;
    int nt   = (i >> 9) & 3;
    int kt   = i >> 11;
    int k = kt * 32 + (lane >> 4) * 8 + (j & 7) + (j >> 3) * 16;
    int n = nt * 16 + (lane & 15);
    sB[i] = f2bf(W[k * DIM + n]);
  }
}

__device__ __forceinline__ v16bf load_b_frag(const __bf16* sB, int kt, int nt, int lane) {
  return *(const v16bf*)(sB + ((kt * 4 + nt) * 32 + lane) * 16);
}

__device__ __forceinline__ v8f wmma_bf16(v16bf a, v16bf b, v8f c) {
  return __builtin_amdgcn_wmma_f32_16x16x32_bf16(false, a, false, b, (short)0, c, false, false);
}

__device__ __forceinline__ float relu(float v) { return v > 0.f ? v : 0.f; }

// wave-uniform tile index, proven uniform to the compiler (readfirstlane is
// identity here since threadIdx.x>>5 is constant within a wave32)
__device__ __forceinline__ int wave_tile() {
  return __builtin_amdgcn_readfirstlane(blockIdx.x * 8 + (threadIdx.x >> 5));
}

// ---------------- init: h0 = h = relu([x_src || e_feat] @ W_init + b) ----------------
__global__ void __launch_bounds__(256) k_init(
    const float* __restrict__ node_feature, const float* __restrict__ edge_feature,
    const int* __restrict__ edge_src,
    const float* __restrict__ W_init, const float* __restrict__ b_init,
    float* __restrict__ h0, float* __restrict__ h, int E) {
  __shared__ alignas(32) __bf16 sB[4 * 4 * 32 * 16];  // 16 KB
  stage_w<4>(W_init, sB);
  __syncthreads();

  int lane = threadIdx.x & 31;
  int r = lane & 15, half = lane >> 4;
  int tile = wave_tile();                 // SGPR
  int nTiles = (E + 15) >> 4;
  if (tile >= nTiles) return;             // scalar branch; EXEC all-ones for WMMA

  long tileBase = (long)tile * 16;        // SGPR pair
  int  rmax = (int)(E - tileBase);        // SGPR
  bool full = rmax >= 16;                 // scalar condition

  long eL = tileBase + (r < rmax ? r : rmax - 1);
  int src = edge_src[eL];

  // bias values for all 4 column tiles, loaded up front
  float bias0 = b_init[r], bias1 = b_init[r + 16], bias2 = b_init[r + 32], bias3 = b_init[r + 48];
  float biasv[4] = {bias0, bias1, bias2, bias3};

  v16bf a0 = make_a_frag(node_feature + (long)src * DIM, 0,  half);
  v16bf a1 = make_a_frag(node_feature + (long)src * DIM, 32, half);
  v16bf a2 = make_a_frag(edge_feature + eL * DIM, 0,  half);
  v16bf a3 = make_a_frag(edge_feature + eL * DIM, 32, half);

  float* h0p = h0 + tileBase * DIM;       // SGPR base -> saddr stores
  float* hp  = h  + tileBase * DIM;

#pragma unroll
  for (int nt = 0; nt < 4; ++nt) {
    v16bf b0 = load_b_frag(sB, 0, nt, lane);
    v16bf b1 = load_b_frag(sB, 1, nt, lane);
    v16bf b2 = load_b_frag(sB, 2, nt, lane);
    v16bf b3 = load_b_frag(sB, 3, nt, lane);
    v8f c = {};
    c = wmma_bf16(a0, b0, c);
    c = wmma_bf16(a1, b1, c);
    c = wmma_bf16(a2, b2, c);
    c = wmma_bf16(a3, b3, c);
    int col = nt * 16 + r;
    float bias = biasv[nt];
    unsigned boff = (unsigned)(((half * 8) * DIM + col) * 4);  // 32-bit byte offset
    if (full) {
#pragma unroll
      for (int i = 0; i < 8; ++i) {
        float v = relu(c[i] + bias);
        st_off(h0p, boff + i * (DIM * 4), v);
        st_off(hp,  boff + i * (DIM * 4), v);
      }
    } else {
#pragma unroll
      for (int i = 0; i < 8; ++i) {
        if (half * 8 + i < rmax) {
          float v = relu(c[i] + bias);
          st_off(h0p, boff + i * (DIM * 4), v);
          st_off(hp,  boff + i * (DIM * 4), v);
        }
      }
    }
  }
}

// ---------------- agg = 0 ----------------
__global__ void __launch_bounds__(256) k_zero(float* __restrict__ p, long n) {
  long i = ((long)blockIdx.x * 256 + threadIdx.x) * 4;
  if (i < n) *(float4*)(p + i) = make_float4(0.f, 0.f, 0.f, 0.f);
}

// ---------------- agg[dst[e]] += h[e]  (segment_sum via L2-resident f32 atomics) ----------------
__global__ void __launch_bounds__(256) k_scatter(
    const float* __restrict__ h, const int* __restrict__ dst,
    float* __restrict__ agg, int E) {
  long t = (long)blockIdx.x * 256 + threadIdx.x;
  long total = (long)E * 16;
  if (t >= total) return;
  long e = t >> 4;
  int  q = (int)(t & 15);
  float4 v = *(const float4*)(h + e * DIM + q * 4);
  int d = dst[e];
  float* base = agg + (long)d * DIM + q * 4;
  atomicAdd(base + 0, v.x);
  atomicAdd(base + 1, v.y);
  atomicAdd(base + 2, v.z);
  atomicAdd(base + 3, v.w);
}

// ---------------- h = relu(h0 + (agg[src] - h[rev]) @ W_t + b_t), in place ----------------
// rev_index = e^1 stays inside the aligned 16-edge tile owned by this wave: the wave
// reads all of its tile's h into registers before any store, so in-place is safe.
__global__ void __launch_bounds__(256) k_step(
    const float* __restrict__ agg, const int* __restrict__ edge_src,
    const int* __restrict__ rev_index,
    const float* __restrict__ W, const float* __restrict__ bvec,
    const float* __restrict__ h0, float* __restrict__ h, int E) {
  __shared__ alignas(32) __bf16 sB[2 * 4 * 32 * 16];  // 8 KB
  stage_w<2>(W, sB);
  __syncthreads();

  int lane = threadIdx.x & 31;
  int r = lane & 15, half = lane >> 4;
  int tile = wave_tile();
  int nTiles = (E + 15) >> 4;
  if (tile >= nTiles) return;

  long tileBase = (long)tile * 16;
  int  rmax = (int)(E - tileBase);
  bool full = rmax >= 16;

  long eL = tileBase + (r < rmax ? r : rmax - 1);
  int src = edge_src[eL];
  int rv  = rev_index[eL];

  const float* h0p = h0 + tileBase * DIM;
  float*       hp  = h  + tileBase * DIM;

  // pull the h0 rows this wave adds in the epilogue toward the WGP
  __builtin_prefetch(h0p + lane * 32, 0, 3);

  float biasv[4] = {bvec[r], bvec[r + 16], bvec[r + 32], bvec[r + 48]};

  const float* ar = agg + (long)src * DIM;
  const float* hr = h   + (long)rv  * DIM;
  v16bf a0 = make_m_frag(ar, hr, 0,  half);
  v16bf a1 = make_m_frag(ar, hr, 32, half);

#pragma unroll
  for (int nt = 0; nt < 4; ++nt) {
    v16bf b0 = load_b_frag(sB, 0, nt, lane);
    v16bf b1 = load_b_frag(sB, 1, nt, lane);
    v8f c = {};
    c = wmma_bf16(a0, b0, c);
    c = wmma_bf16(a1, b1, c);
    int col = nt * 16 + r;
    float bias = biasv[nt];
    unsigned boff = (unsigned)(((half * 8) * DIM + col) * 4);
    if (full) {
#pragma unroll
      for (int i = 0; i < 8; ++i) {
        float v = relu(ld_off(h0p, boff + i * (DIM * 4)) + c[i] + bias);
        st_off(hp, boff + i * (DIM * 4), v);
      }
    } else {
#pragma unroll
      for (int i = 0; i < 8; ++i) {
        if (half * 8 + i < rmax) {
          float v = relu(ld_off(h0p, boff + i * (DIM * 4)) + c[i] + bias);
          st_off(hp, boff + i * (DIM * 4), v);
        }
      }
    }
  }
}

// ---------------- node_out = relu([node_feature || node_agg] @ W_proj + b) ----------------
__global__ void __launch_bounds__(256) k_proj(
    const float* __restrict__ node_feature, const float* __restrict__ agg,
    const float* __restrict__ W, const float* __restrict__ bvec,
    float* __restrict__ out, int N) {
  __shared__ alignas(32) __bf16 sB[4 * 4 * 32 * 16];  // 16 KB
  stage_w<4>(W, sB);
  __syncthreads();

  int lane = threadIdx.x & 31;
  int r = lane & 15, half = lane >> 4;
  int tile = wave_tile();
  int nTiles = (N + 15) >> 4;
  if (tile >= nTiles) return;

  long tileBase = (long)tile * 16;
  int  rmax = (int)(N - tileBase);
  bool full = rmax >= 16;

  long nL = tileBase + (r < rmax ? r : rmax - 1);

  float biasv[4] = {bvec[r], bvec[r + 16], bvec[r + 32], bvec[r + 48]};

  v16bf a0 = make_a_frag(node_feature + nL * DIM, 0,  half);
  v16bf a1 = make_a_frag(node_feature + nL * DIM, 32, half);
  v16bf a2 = make_a_frag(agg          + nL * DIM, 0,  half);
  v16bf a3 = make_a_frag(agg          + nL * DIM, 32, half);

  float* op = out + tileBase * DIM;

#pragma unroll
  for (int nt = 0; nt < 4; ++nt) {
    v16bf b0 = load_b_frag(sB, 0, nt, lane);
    v16bf b1 = load_b_frag(sB, 1, nt, lane);
    v16bf b2 = load_b_frag(sB, 2, nt, lane);
    v16bf b3 = load_b_frag(sB, 3, nt, lane);
    v8f c = {};
    c = wmma_bf16(a0, b0, c);
    c = wmma_bf16(a1, b1, c);
    c = wmma_bf16(a2, b2, c);
    c = wmma_bf16(a3, b3, c);
    int col = nt * 16 + r;
    float bias = biasv[nt];
    unsigned boff = (unsigned)(((half * 8) * DIM + col) * 4);
    if (full) {
#pragma unroll
      for (int i = 0; i < 8; ++i) st_off(op, boff + i * (DIM * 4), relu(c[i] + bias));
    } else {
#pragma unroll
      for (int i = 0; i < 8; ++i)
        if (half * 8 + i < rmax) st_off(op, boff + i * (DIM * 4), relu(c[i] + bias));
    }
  }
}

extern "C" void kernel_launch(void* const* d_in, const int* in_sizes, int n_in,
                              void* d_out, int out_size, void* d_ws, size_t ws_size,
                              hipStream_t stream) {
  const float* node_feature = (const float*)d_in[0];
  const float* edge_feature = (const float*)d_in[1];
  const int*   edge_src     = (const int*)d_in[2];
  const int*   edge_dst     = (const int*)d_in[3];
  const int*   rev_index    = (const int*)d_in[4];
  const float* W_init       = (const float*)d_in[5];
  const float* b_init       = (const float*)d_in[6];
  const float* W_steps      = (const float*)d_in[7];
  const float* b_steps      = (const float*)d_in[8];
  const float* W_proj       = (const float*)d_in[9];
  const float* b_proj       = (const float*)d_in[10];

  const int N     = in_sizes[0] / DIM;
  const int E     = in_sizes[1] / DIM;
  const int STEPS = in_sizes[7] / (DIM * DIM);

  float* node_out = (float*)d_out;                   // [N, DIM]
  float* h        = (float*)d_out + (long)N * DIM;   // [E, DIM], updated in place
  float* agg      = (float*)d_ws;                    // [N, DIM]
  float* h0       = (float*)d_ws + (long)N * DIM;    // [E, DIM]

  const int eTiles  = (E + 15) / 16;
  const int eBlocks = (eTiles + 7) / 8;
  const int nTiles  = (N + 15) / 16;
  const int nBlocks = (nTiles + 7) / 8;
  const long aggElems   = (long)N * DIM;
  const int  zeroBlocks = (int)((aggElems / 4 + 255) / 256);
  const int  scatBlocks = (int)(((long)E * 16 + 255) / 256);

  k_init<<<eBlocks, 256, 0, stream>>>(node_feature, edge_feature, edge_src,
                                      W_init, b_init, h0, h, E);
  for (int t = 0; t < STEPS; ++t) {
    k_zero<<<zeroBlocks, 256, 0, stream>>>(agg, aggElems);
    k_scatter<<<scatBlocks, 256, 0, stream>>>(h, edge_dst, agg, E);
    k_step<<<eBlocks, 256, 0, stream>>>(agg, edge_src, rev_index,
                                        W_steps + (long)t * DIM * DIM,
                                        b_steps + (long)t * DIM, h0, h, E);
  }
  k_zero<<<zeroBlocks, 256, 0, stream>>>(agg, aggElems);
  k_scatter<<<scatBlocks, 256, 0, stream>>>(h, edge_dst, agg, E);
  k_proj<<<nBlocks, 256, 0, stream>>>(node_feature, agg, W_proj, b_proj, node_out, N);
}
